// VAE_9938554323034
// MI455X (gfx1250) — compile-verified, matrix operand order checked by reading, not run on hardware
//
#include <hip/hip_runtime.h>
#include <hip/hip_bf16.h>
#include <stdint.h>

// ---------------------------------------------------------------------------
// Types for CDNA5 WMMA (wave32): v_wmma_f32_16x16x32_bf16
// ---------------------------------------------------------------------------
typedef __bf16 bf16_t;
typedef __attribute__((ext_vector_type(8)))  __bf16 v8bf;
typedef __attribute__((ext_vector_type(16))) __bf16 v16bf;
typedef __attribute__((ext_vector_type(8)))  float  v8f;
typedef __attribute__((ext_vector_type(4)))  int    v4i;

#define GAS __attribute__((address_space(1)))
#define LAS __attribute__((address_space(3)))

// Async global->LDS path (ASYNCcnt-tracked): param0 = AS1 int4* (global src),
// param1 = LDS dst (signature confirmed by round-2 diagnostics).
#if defined(__has_builtin)
#  if __has_builtin(__builtin_amdgcn_global_load_async_to_lds_b128)
#    define USE_ASYNC_LDS 1
#  endif
#endif
#ifndef USE_ASYNC_LDS
#  define USE_ASYNC_LDS 0
#endif

__device__ __forceinline__ void wait_async0() {
#if defined(__has_builtin) && __has_builtin(__builtin_amdgcn_s_wait_asynccnt)
    __builtin_amdgcn_s_wait_asynccnt(0);
#else
    asm volatile("s_wait_asynccnt 0x0" ::: "memory");
#endif
}

// Per the ISA 16-bit A/B fragment layout:
//  lane l<16 : K = {koff..koff+7} in v0..3, {koff+16..koff+23} in v4..7, koff=0
//  lane l>=16: same with koff=8
// For K-contiguous data each lane performs two 16B loads 32B apart.  The same
// helper serves global memory and LDS (addrspace inferred after inlining).
__device__ __forceinline__ v16bf load_frag16(const bf16_t* p) {
    v8bf lo = *reinterpret_cast<const v8bf*>(p);
    v8bf hi = *reinterpret_cast<const v8bf*>(p + 16);
    return __builtin_shufflevector(lo, hi, 0,1,2,3,4,5,6,7,8,9,10,11,12,13,14,15);
}

__device__ __forceinline__ v8f wmma_bf16(v16bf a, v16bf b, v8f c) {
    return __builtin_amdgcn_wmma_f32_16x16x32_bf16(false, a, false, b,
                                                   (short)0, c, false, false);
}

// ---------------------------------------------------------------------------
// C[M,N](f32) = A[M,K](bf16, row-major, lda) * W[N,K](bf16, row-major, ldw)^T
//             (+ bias[n]);  optional bf16 mirror Cbf.
//
// Block: 256 threads = 8 waves; block tile 128M x 128N; wave tile 32M x 64N.
// M multiple of 128 (M = 512 everywhere), K multiple of 128 (padded) so the
// 64-K chunk count is even and the pipeline can ping-pong without register
// rotation (no aCur<-aNext copies, no WMMA->VALU hazard nops).
// W padded with zero rows to a multiple of 128 -> only the store epilogue
// needs an n < N guard; EXEC stays all-ones through the WMMAs.
//
// B tile is double-buffered through LDS (row stride 72 bf16 = 144B so the 16
// fragment rows land in distinct bank groups).  While chunk c is consumed,
// chunk c+1's B tile is in flight (async-to-LDS, ASYNCcnt) and chunk c+1's A
// fragments are prefetched into the alternate register set.
// ---------------------------------------------------------------------------
#define KC 64
#define LDSTR (KC + 8)   // 72 bf16 per LDS row

__global__ __launch_bounds__(256) void wmma_gemm_bf16(
    const bf16_t* __restrict__ A, int lda,
    const bf16_t* __restrict__ W, int ldw,
    float* __restrict__ C, int ldc,
    bf16_t* __restrict__ Cbf, int ldcb,
    const float* __restrict__ bias,
    int N, int K)
{
    __shared__ __align__(16) bf16_t ldsB[2][128][LDSTR];

    const int tid  = threadIdx.x;
    const int lane = tid & 31;
    const int wave = tid >> 5;
    const int wm   = wave & 3;            // 4 waves along M
    const int wn   = wave >> 2;           // 2 waves along N
    const int lr   = lane & 15;
    const int lh   = lane >> 4;
    const int koff = lh * 8;

    const int mBase = blockIdx.y * 128 + wm * 32;
    const int nBlk  = blockIdx.x * 128;
    const int nBase = nBlk + wn * 64;

    // cooperative B loader: thread covers row = tid/2, k-half = (tid&1)*32
    const int brow = tid >> 1;
    const int bk   = (tid & 1) * 32;
    const bf16_t* gB = W + (size_t)(nBlk + brow) * ldw + bk;
    bf16_t* lB0 = &ldsB[0][brow][bk];
    bf16_t* lB1 = &ldsB[1][brow][bk];

    // per-wave LDS fragment base pointers (hoisted out of the loop)
    const bf16_t* lbase0 = &ldsB[0][wn * 64 + lr][koff];
    const bf16_t* lbase1 = &ldsB[1][wn * 64 + lr][koff];

    const bf16_t* pa = A + (size_t)(mBase + lr) * lda + koff;
    const size_t aRow16 = (size_t)16 * lda;

    v8f acc[2][4];
#pragma unroll
    for (int i = 0; i < 2; ++i)
#pragma unroll
        for (int j = 0; j < 4; ++j)
            acc[i][j] = (v8f){0.f,0.f,0.f,0.f,0.f,0.f,0.f,0.f};

    // stage B chunk c into the given LDS buffer (this thread's 64B slice)
    auto issueB = [&](int c, bf16_t* ldst) {
        const bf16_t* g = gB + (size_t)c * KC;
#if USE_ASYNC_LDS
#pragma unroll
        for (int j = 0; j < 4; ++j)
            __builtin_amdgcn_global_load_async_to_lds_b128(
                (GAS v4i*)(g + j * 8), (LAS v4i*)(ldst + j * 8), 0, 0);
#else
#pragma unroll
        for (int j = 0; j < 4; ++j) {
            v8bf t = *reinterpret_cast<const v8bf*>(g + j * 8);
            *reinterpret_cast<v8bf*>(ldst + j * 8) = t;
        }
#endif
    };
    auto waitB = [&]() {
#if USE_ASYNC_LDS
        wait_async0();
#endif
    };
    // load this wave's A fragments for chunk c : af[substep][mi]
    auto loadA = [&](int c, v16bf (&af)[2][2]) {
        const bf16_t* pan = pa + (size_t)c * KC;
#pragma unroll
        for (int s = 0; s < 2; ++s)
#pragma unroll
            for (int i = 0; i < 2; ++i)
                af[s][i] = load_frag16(pan + (size_t)i * aRow16 + s * 32);
    };
    // 2 substeps x 8 WMMA against one LDS buffer
    auto compute = [&](const bf16_t* lbase, v16bf (&af)[2][2]) {
#pragma unroll
        for (int s = 0; s < 2; ++s) {
            const bf16_t* lb = lbase + s * 32;
            v16bf b0 = load_frag16(lb);
            v16bf b1 = load_frag16(lb + 16 * LDSTR);
            v16bf b2 = load_frag16(lb + 32 * LDSTR);
            v16bf b3 = load_frag16(lb + 48 * LDSTR);
            acc[0][0] = wmma_bf16(af[s][0], b0, acc[0][0]);
            acc[0][1] = wmma_bf16(af[s][0], b1, acc[0][1]);
            acc[0][2] = wmma_bf16(af[s][0], b2, acc[0][2]);
            acc[0][3] = wmma_bf16(af[s][0], b3, acc[0][3]);
            acc[1][0] = wmma_bf16(af[s][1], b0, acc[1][0]);
            acc[1][1] = wmma_bf16(af[s][1], b1, acc[1][1]);
            acc[1][2] = wmma_bf16(af[s][1], b2, acc[1][2]);
            acc[1][3] = wmma_bf16(af[s][1], b3, acc[1][3]);
        }
    };

    const int chunks = K / KC;   // even by construction (K % 128 == 0)
    v16bf aA[2][2], aB[2][2];

    // ---- prologue: chunk 0 staged + A fragments loaded
    issueB(0, lB0);
    loadA(0, aA);
    waitB();
    __syncthreads();

    for (int c = 0; c < chunks; c += 2) {
        // ---- even chunk from buf0/aA; prefetch c+1 into buf1/aB
        issueB(c + 1, lB1);
        loadA(c + 1, aB);
        compute(lbase0, aA);
        waitB();
        __syncthreads();

        // ---- odd chunk from buf1/aB; prefetch c+2 into buf0/aA
        const bool more = (c + 2) < chunks;
        if (more) {
            issueB(c + 2, lB0);
            loadA(c + 2, aA);
        }
        compute(lbase1, aB);
        if (more) {
            waitB();
            __syncthreads();
        }
    }

    // C/D 16x16 f32 layout: lane 0-15 -> (M = r, N = lane); lane 16-31 -> (M = 8+r).
#pragma unroll
    for (int ti = 0; ti < 2; ++ti) {
#pragma unroll
        for (int tj = 0; tj < 4; ++tj) {
            int n = nBase + tj * 16 + lr;
            if (n < N) {
                float bv = bias ? bias[n] : 0.0f;
#pragma unroll
                for (int r = 0; r < 8; ++r) {
                    int m = mBase + ti * 16 + lh * 8 + r;
                    float val = acc[ti][tj][r] + bv;
                    C[(size_t)m * ldc + n] = val;
                    if (Cbf) Cbf[(size_t)m * ldcb + n] = (bf16_t)val;
                }
            }
        }
    }
}

// ---------------------------------------------------------------------------
// f32 -> bf16 with zero padding: dst is [rowsPad, ldd]; src is [rows, cols].
// ---------------------------------------------------------------------------
__global__ void cvt_pad_bf16(const float* __restrict__ src, bf16_t* __restrict__ dst,
                             int rows, int cols, int ldd, long total)
{
    long idx = (long)blockIdx.x * blockDim.x + threadIdx.x;
    if (idx >= total) return;
    int r = (int)(idx / ldd);
    int c = (int)(idx % ldd);
    float v = (r < rows && c < cols) ? src[(size_t)r * cols + c] : 0.0f;
    dst[idx] = (bf16_t)v;
}

__global__ void zero_u32(uint32_t* __restrict__ p, long n)
{
    long i = (long)blockIdx.x * blockDim.x + threadIdx.x;
    if (i < n) p[i] = 0u;
}

// ---------------------------------------------------------------------------
// Fused GRU gate math.  Gi = x@Wih^T, Gh = h@Whh^T (each [B, 3H], H = 2048).
//   r = sig(i_r + h_r); z = sig(i_z + h_z); n = tanh(i_n + r*h_n)
//   h' = (1-z)*n + z*h   (written f32 + bf16 for the next step's GEMM)
// ---------------------------------------------------------------------------
__global__ void gru_update(const float* __restrict__ Gi, const float* __restrict__ Gh,
                           const float* __restrict__ bi, const float* __restrict__ bh,
                           float* __restrict__ h, bf16_t* __restrict__ hbf)
{
    int idx = blockIdx.x * blockDim.x + threadIdx.x;   // exactly B*H = 1,048,576
    int b = idx >> 11;        // / 2048
    int j = idx & 2047;
    size_t base = (size_t)b * 6144;
    float ir = Gi[base + j]        + bi[j];
    float iz = Gi[base + 2048 + j] + bi[2048 + j];
    float in = Gi[base + 4096 + j] + bi[4096 + j];
    float hr = Gh[base + j]        + bh[j];
    float hz = Gh[base + 2048 + j] + bh[2048 + j];
    float hn = Gh[base + 4096 + j] + bh[4096 + j];
    float r = 1.0f / (1.0f + expf(-(ir + hr)));
    float z = 1.0f / (1.0f + expf(-(iz + hz)));
    float n = tanhf(in + r * hn);
    float hnew = (1.0f - z) * n + z * h[idx];
    h[idx]   = hnew;
    hbf[idx] = (bf16_t)hnew;
}

// ---------------------------------------------------------------------------
// Reparameterize + concat genre:  z[b, c<1019] = mu + exp(0.5*logvar)*eps
//                                 z[b, 1019..1023] = genre[b, 0..4]
// ---------------------------------------------------------------------------
__global__ void build_z(const float* __restrict__ mu, const float* __restrict__ lv,
                        const float* __restrict__ eps, const float* __restrict__ genre,
                        bf16_t* __restrict__ zb)
{
    int idx = blockIdx.x * blockDim.x + threadIdx.x;   // exactly 512*1024
    int b = idx >> 10;
    int c = idx & 1023;
    float v;
    if (c < 1019) {
        size_t o = (size_t)b * 1019 + c;
        v = mu[o] + expf(0.5f * lv[o]) * eps[o];
    } else {
        v = genre[b * 5 + (c - 1019)];
    }
    zb[idx] = (bf16_t)v;
}

// ---------------------------------------------------------------------------
// Host orchestration
// ---------------------------------------------------------------------------
extern "C" void kernel_launch(void* const* d_in, const int* in_sizes, int n_in,
                              void* d_out, int out_size, void* d_ws, size_t ws_size,
                              hipStream_t stream)
{
    (void)in_sizes; (void)n_in; (void)out_size; (void)ws_size;

    constexpr int T = 256, B = 512, IN = 512, XIN = 517, XINP = 640;  // K pad -> mult of 128
    constexpr int H = 2048, G3 = 3 * H, LATM = 1019, LAT = 1024;

    const float* x        = (const float*)d_in[0];
    const float* genre    = (const float*)d_in[1];
    const float* eps      = (const float*)d_in[2];
    const float* enc_w_ih = (const float*)d_in[3];
    const float* enc_w_hh = (const float*)d_in[4];
    const float* enc_b_ih = (const float*)d_in[5];
    const float* enc_b_hh = (const float*)d_in[6];
    const float* dec_w_ih = (const float*)d_in[7];
    const float* dec_w_hh = (const float*)d_in[8];
    const float* dec_b_ih = (const float*)d_in[9];
    const float* dec_b_hh = (const float*)d_in[10];
    const float* fc_mu_w  = (const float*)d_in[11];
    const float* fc_mu_b  = (const float*)d_in[12];
    const float* fc_lv_w  = (const float*)d_in[13];
    const float* fc_lv_b  = (const float*)d_in[14];
    const float* fc_out_w = (const float*)d_in[15];
    const float* fc_out_b = (const float*)d_in[16];
    const float* fc_dec_w = (const float*)d_in[17];
    const float* fc_dec_b = (const float*)d_in[18];

    float* out = (float*)d_out;
    const size_t MU_OFF = (size_t)T * B * IN;            // 67,108,864
    const size_t LV_OFF = MU_OFF + (size_t)B * LATM;     // + 521,728

    // ---- workspace carve-out (256B aligned) ----
    uint8_t* ws = (uint8_t*)d_ws;
    size_t off = 0;
    auto alloc = [&](size_t bytes) -> void* {
        off = (off + 255) & ~(size_t)255;
        void* p = ws + off;
        off += bytes;
        return p;
    };
    bf16_t* wih_e = (bf16_t*)alloc((size_t)G3 * XINP * 2);   // enc W_ih  [6144,640]
    bf16_t* whh_e = (bf16_t*)alloc((size_t)G3 * H * 2);      // enc W_hh  [6144,2048]
    bf16_t* wih_d = (bf16_t*)alloc((size_t)G3 * IN * 2);     // dec W_ih  [6144,512]
    bf16_t* whh_d = (bf16_t*)alloc((size_t)G3 * H * 2);      // dec W_hh  [6144,2048]
    bf16_t* wmu   = (bf16_t*)alloc((size_t)LAT * H * 2);     // fc_mu     [1024,2048]
    bf16_t* wlv   = (bf16_t*)alloc((size_t)LAT * H * 2);     // fc_logvar [1024,2048]
    bf16_t* wout  = (bf16_t*)alloc((size_t)H * LAT * 2);     // fc_output [2048,1024]
    bf16_t* wdec  = (bf16_t*)alloc((size_t)IN * H * 2);      // fc_decoder[512,2048]
    bf16_t* xbf   = (bf16_t*)alloc((size_t)B * XINP * 2);    // x_t bf16  [512,640]
    bf16_t* hbf   = (bf16_t*)alloc((size_t)B * H * 2);       // h  bf16   [512,2048]
    bf16_t* zbf   = (bf16_t*)alloc((size_t)B * LAT * 2);     // z  bf16   [512,1024]
    bf16_t* nbf   = (bf16_t*)alloc((size_t)B * IN * 2);      // note bf16 [512,512]
    float*  Gi    = (float*) alloc((size_t)B * G3 * 4);      // input gates
    float*  Gh    = (float*) alloc((size_t)B * G3 * 4);      // hidden gates
    float*  hfp   = (float*) alloc((size_t)B * H * 4);       // h f32

    auto cvt = [&](const float* src, bf16_t* dst, int rows, int cols, int ldd, int rowsPad) {
        long total = (long)rowsPad * ldd;
        cvt_pad_bf16<<<dim3((unsigned)((total + 255) / 256)), dim3(256), 0, stream>>>(
            src, dst, rows, cols, ldd, total);
    };
    auto zero = [&](void* p, size_t bytes) {
        long n = (long)(bytes / 4);
        zero_u32<<<dim3((unsigned)((n + 255) / 256)), dim3(256), 0, stream>>>((uint32_t*)p, n);
    };
    auto gemm = [&](const bf16_t* A, int lda, const bf16_t* W, int ldw,
                    float* C, int ldc, bf16_t* Cbf, int ldcb,
                    const float* bias, int N, int K) {
        dim3 grid((unsigned)((N + 127) / 128), (unsigned)(B / 128));
        wmma_gemm_bf16<<<grid, dim3(256), 0, stream>>>(A, lda, W, ldw, C, ldc,
                                                       Cbf, ldcb, bias, N, K);
    };

    // ---- weight conversion (bf16, K padded to mult-of-128, N padded to mult-of-128)
    cvt(enc_w_ih, wih_e, G3, XIN, XINP, G3);
    cvt(enc_w_hh, whh_e, G3, H, H, G3);
    cvt(dec_w_ih, wih_d, G3, IN, IN, G3);
    cvt(dec_w_hh, whh_d, G3, H, H, G3);
    cvt(fc_mu_w,  wmu, LATM, H, H, LAT);
    cvt(fc_lv_w,  wlv, LATM, H, H, LAT);
    cvt(fc_out_w, wout, H, LAT, LAT, H);
    cvt(fc_dec_w, wdec, IN, H, H, IN);

    // ---- h0 = 0
    zero(hfp, (size_t)B * H * 4);
    zero(hbf, (size_t)B * H * 2);

    // ---- encoder: 256 sequential GRU steps
    for (int t = 0; t < T; ++t) {
        cvt(x + (size_t)t * B * XIN, xbf, B, XIN, XINP, B);
        gemm(xbf, XINP, wih_e, XINP, Gi, G3, nullptr, 0, nullptr, G3, XINP);
        gemm(hbf, H,    whh_e, H,    Gh, G3, nullptr, 0, nullptr, G3, H);
        gru_update<<<dim3((B * H) / 256), dim3(256), 0, stream>>>(
            Gi, Gh, enc_b_ih, enc_b_hh, hfp, hbf);
    }

    // ---- latent: mu / logvar straight into d_out, then z and decoder h0
    gemm(hbf, H, wmu, H, out + MU_OFF, LATM, nullptr, 0, fc_mu_b, LATM, H);
    gemm(hbf, H, wlv, H, out + LV_OFF, LATM, nullptr, 0, fc_lv_b, LATM, H);
    build_z<<<dim3((B * LAT) / 256), dim3(256), 0, stream>>>(
        out + MU_OFF, out + LV_OFF, eps, genre, zbf);
    gemm(zbf, LAT, wout, LAT, hfp, H, hbf, H, fc_out_b, H, LAT);

    // ---- decoder: note0 = 0, then 256 autoregressive steps
    zero(nbf, (size_t)B * IN * 2);
    for (int t = 0; t < T; ++t) {
        gemm(nbf, IN, wih_d, IN, Gi, G3, nullptr, 0, nullptr, G3, IN);
        gemm(hbf, H,  whh_d, H,  Gh, G3, nullptr, 0, nullptr, G3, H);
        gru_update<<<dim3((B * H) / 256), dim3(256), 0, stream>>>(
            Gi, Gh, dec_b_ih, dec_b_hh, hfp, hbf);
        gemm(hbf, H, wdec, H, out + (size_t)t * B * IN, IN, nbf, IN,
             fc_dec_b, IN, H);
    }
}
